// Conv3dAsLinear_6021544149251
// MI455X (gfx1250) — compile-verified, matrix operand order checked by reading, not run on hardware
//
#include <hip/hip_runtime.h>
#include <hip/hip_bf16.h>
#include <stdint.h>

// ---- CDNA5 WMMA vector types ----
typedef __bf16    bf16x16 __attribute__((ext_vector_type(16)));
typedef float     f32x8   __attribute__((ext_vector_type(8)));
typedef uint32_t  u32x8   __attribute__((ext_vector_type(8)));
typedef uint32_t  u32x4   __attribute__((ext_vector_type(4)));

#define B_    2
#define CI_   16
#define T_    32
#define H_    128
#define W_    128
#define CO_   32
#define THW_  (T_ * H_ * W_)              // 524288
#define PT_   (T_ + 2)                    // 34  (halo-padded)
#define PH_   (H_ + 2)                    // 130
#define PW_   (W_ + 2)                    // 130
#define PPIX  (B_ * PT_ * PH_ * PW_)      // 1,149,200 padded pixels
#define NTAP  27
#define NTAPP 28
#define NCHUNK 14                         // 28 taps / 2 taps per K=32 chunk

#define WLDS_USHORTS (2 * NTAPP * CO_ * CI_)   // hi+lo bf16 weights = 57344 B

// ============================================================================
// Pass 1: split f32 input into HALO-PADDED channel-last bf16 hi/lo planes:
//   plane[((b*34 + t+1)*130 + h+1)*130 + (w+1)][ci], zero in the halo.
//   hi = top16(x) (truncation), lo = bf16(x - hi)  -> combined ~2^-17 rel.
// Padding makes every conv tap load unconditional in pass 2.
// ============================================================================
__global__ __launch_bounds__(256) void split_pad_bf16_kernel(
    const float* __restrict__ X,            // [B][CI][T][H][W]
    unsigned short* __restrict__ WShi,      // [PPIX][CI]
    unsigned short* __restrict__ WSlo)      // [PPIX][CI]
{
    const long p = (long)blockIdx.x * 256 + threadIdx.x;    // padded pixel id
    if (p >= PPIX) return;

    int  wp = (int)(p % PW_);
    long r  = p / PW_;
    int  hp = (int)(r % PH_);  r /= PH_;
    int  tp = (int)(r % PT_);
    int  b  = (int)(r / PT_);

    u32x8 hw = {0,0,0,0,0,0,0,0};
    u32x8 lw = {0,0,0,0,0,0,0,0};

    const bool interior = (wp >= 1) && (wp <= W_) && (hp >= 1) && (hp <= H_) &&
                          (tp >= 1) && (tp <= T_);
    if (interior) {
        const float* src = X + (long)b * CI_ * THW_ +
                           ((long)(tp - 1) * H_ + (hp - 1)) * W_ + (wp - 1);
        #pragma unroll
        for (int q = 0; q < 8; ++q) {
            float f0 = src[(long)(2 * q + 0) * THW_];
            float f1 = src[(long)(2 * q + 1) * THW_];
            uint32_t u0 = __float_as_uint(f0), u1 = __float_as_uint(f1);
            float l0 = f0 - __uint_as_float(u0 & 0xFFFF0000u);   // exact residual
            float l1 = f1 - __uint_as_float(u1 & 0xFFFF0000u);
            hw[q] = (u0 >> 16) | (u1 & 0xFFFF0000u);
            lw[q] = (__float_as_uint(l0) >> 16) | (__float_as_uint(l1) & 0xFFFF0000u);
        }
    }
    *(u32x8*)(WShi + p * CI_) = hw;   // 32B contiguous store per lane
    *(u32x8*)(WSlo + p * CI_) = lw;
}

// ============================================================================
// Pass 2: implicit-GEMM conv via V_WMMA_F32_16X16X32_BF16, bf16x2 split.
// One WG (8 wave32) per (b,t,h) row: M=128 W-pixels x N=32 Cout.
// Halo padding => all 27 tap loads are unconditional b128s at COMPILE-TIME
// immediate offsets (max ~1.09MB, fits signed 24-bit VMEM offset) from one
// per-lane base pointer: zero VALU address math in the hot loop.
// ============================================================================
__global__ __launch_bounds__(256) void conv3d_wmma_ws_kernel(
    const unsigned short* __restrict__ WShi,
    const unsigned short* __restrict__ WSlo,
    const float* __restrict__ Wt,     // [CO][CI][3][3][3]
    const float* __restrict__ Bias,   // [CO]
    float* __restrict__ Out)          // [B][CO][T][H][W]
{
    __shared__ __align__(16) unsigned char smem[WLDS_USHORTS * 2];
    unsigned short* Wl = (unsigned short*)smem;
    float*          Ot = (float*)smem;

    const int tid  = threadIdx.x;
    const int wave = tid >> 5;
    const int lane = tid & 31;
    const int g    = (lane >> 4) & 1;
    const int col  = lane & 15;

    const int bh = blockIdx.x;
    const int h  = bh & 127;
    const int t  = (bh >> 7) & 31;
    const int b  = bh >> 12;

    // ---- stage weights -> LDS as bf16 hi/lo, layout [hl][tap][n][ci] ----
    for (int idx = tid; idx < NTAPP * CO_ * CI_; idx += 256) {
        int tap = idx >> 9;
        int n   = (idx >> 4) & 31;
        int ci  = idx & 15;
        float v = (tap < NTAP) ? Wt[n * 432 + ci * 27 + tap] : 0.0f;
        uint32_t u  = __float_as_uint(v);
        float    lf = v - __uint_as_float(u & 0xFFFF0000u);
        Wl[idx] = (unsigned short)(u >> 16);
        Wl[NTAPP * CO_ * CI_ + idx] = (unsigned short)(__float_as_uint(lf) >> 16);
    }
    __syncthreads();

    f32x8 acc0 = {0.f,0.f,0.f,0.f,0.f,0.f,0.f,0.f};
    f32x8 acc1 = {0.f,0.f,0.f,0.f,0.f,0.f,0.f,0.f};

    const int  wbase = wave * 16;
    const int  wpix  = wbase + col;
    // Per-lane base: padded pixel (b, t, h, wpix); tap (kt,kh,kw) adds the
    // constant ((kt*130+kh)*130+kw)*CI_ element offset.
    const long basePix = (((long)b * PT_ + t) * PH_ + h) * PW_ + wpix;
    const unsigned short* pHi = WShi + basePix * CI_ + 8 * g;
    const unsigned short* pLo = WSlo + basePix * CI_ + 8 * g;

    #pragma unroll
    for (int c = 0; c < NCHUNK; ++c) {
        u32x8 ah, al;
        #pragma unroll
        for (int j = 0; j < 2; ++j) {
            const int tap = 2 * c + j;             // literal after unroll
            u32x4 qh, ql;
            if (tap < NTAP) {
                const int kt = tap / 9, kh = (tap % 9) / 3, kw = tap % 3;
                const long d = ((long)(kt * PH_ + kh) * PW_ + kw) * CI_; // const
                qh = *(const u32x4*)(pHi + d);     // immediate-offset b128
                ql = *(const u32x4*)(pLo + d);
            } else {                               // K pad (chunk 13, j=1)
                qh = (u32x4){0u,0u,0u,0u};
                ql = (u32x4){0u,0u,0u,0u};
            }
            #pragma unroll
            for (int q = 0; q < 4; ++q) { ah[j*4+q] = qh[q]; al[j*4+q] = ql[q]; }
        }
        bf16x16 A_hi = __builtin_bit_cast(bf16x16, ah);
        bf16x16 A_lo = __builtin_bit_cast(bf16x16, al);

        const int tap0 = 2 * c, tap1 = 2 * c + 1;
        auto loadB = [&](int hl, int tn) -> bf16x16 {
            int n = tn * 16 + col;
            u32x4 q0 = *(const u32x4*)&Wl[(((hl * NTAPP) + tap0) * CO_ + n) * CI_ + 8 * g];
            u32x4 q1 = *(const u32x4*)&Wl[(((hl * NTAPP) + tap1) * CO_ + n) * CI_ + 8 * g];
            u32x8 bw;
            #pragma unroll
            for (int q = 0; q < 4; ++q) { bw[q] = q0[q]; bw[4+q] = q1[q]; }
            return __builtin_bit_cast(bf16x16, bw);
        };
        bf16x16 Bh0 = loadB(0, 0), Bl0 = loadB(1, 0);
        bf16x16 Bh1 = loadB(0, 1), Bl1 = loadB(1, 1);

        // bf16x2-split matmul: hi*hi + lo*hi + hi*lo (f32 accumulate)
        acc0 = __builtin_amdgcn_wmma_f32_16x16x32_bf16(false, A_hi, false, Bh0, (short)0, acc0, false, false);
        acc0 = __builtin_amdgcn_wmma_f32_16x16x32_bf16(false, A_lo, false, Bh0, (short)0, acc0, false, false);
        acc0 = __builtin_amdgcn_wmma_f32_16x16x32_bf16(false, A_hi, false, Bl0, (short)0, acc0, false, false);
        acc1 = __builtin_amdgcn_wmma_f32_16x16x32_bf16(false, A_hi, false, Bh1, (short)0, acc1, false, false);
        acc1 = __builtin_amdgcn_wmma_f32_16x16x32_bf16(false, A_lo, false, Bh1, (short)0, acc1, false, false);
        acc1 = __builtin_amdgcn_wmma_f32_16x16x32_bf16(false, A_hi, false, Bl1, (short)0, acc1, false, false);
    }

    // ---- transpose D through (reused) LDS for coalesced stores ----
    __syncthreads();
    #pragma unroll
    for (int j = 0; j < 8; ++j) {                  // D: M = j + 8g, N = col
        int w = wbase + j + 8 * g;
        Ot[col * 129 + w]        = acc0[j];
        Ot[(16 + col) * 129 + w] = acc1[j];
    }
    __syncthreads();

    const long oB = (((long)b * CO_) * T_ + t) * (long)(H_ * W_) + (long)h * W_;
    for (int i = tid; i < CO_ * W_; i += 256) {
        int n = i >> 7, w = i & 127;
        float v = Ot[n * 129 + w] + Bias[n];
        __builtin_nontemporal_store(v, &Out[oB + (long)n * THW_ + w]);
    }
}

// ============================================================================
// Fallback (ws too small): single-pass version (in-loop split, masked loads).
// ============================================================================
__global__ __launch_bounds__(256) void conv3d_wmma_fallback_kernel(
    const float* __restrict__ X, const float* __restrict__ Wt,
    const float* __restrict__ Bias, float* __restrict__ Out)
{
    __shared__ __align__(16) unsigned char smem[WLDS_USHORTS * 2];
    unsigned short* Wl = (unsigned short*)smem;
    float*          Ot = (float*)smem;

    const int tid  = threadIdx.x;
    const int wave = tid >> 5;
    const int lane = tid & 31;
    const int g    = (lane >> 4) & 1;
    const int col  = lane & 15;

    const int bh = blockIdx.x;
    const int h  = bh & 127;
    const int t  = (bh >> 7) & 31;
    const int b  = bh >> 12;

    for (int idx = tid; idx < NTAPP * CO_ * CI_; idx += 256) {
        int tap = idx >> 9, n = (idx >> 4) & 31, ci = idx & 15;
        float v = (tap < NTAP) ? Wt[n * 432 + ci * 27 + tap] : 0.0f;
        uint32_t u  = __float_as_uint(v);
        float    lf = v - __uint_as_float(u & 0xFFFF0000u);
        Wl[idx] = (unsigned short)(u >> 16);
        Wl[NTAPP * CO_ * CI_ + idx] = (unsigned short)(__float_as_uint(lf) >> 16);
    }
    __syncthreads();

    f32x8 acc0 = {0.f,0.f,0.f,0.f,0.f,0.f,0.f,0.f};
    f32x8 acc1 = {0.f,0.f,0.f,0.f,0.f,0.f,0.f,0.f};
    const int  wbase = wave * 16, wpix = wbase + col;
    const float* Xb = X + (long)b * CI_ * THW_;

    #pragma unroll
    for (int c = 0; c < NCHUNK; ++c) {
        u32x8 ah, al;
        #pragma unroll
        for (int j = 0; j < 2; ++j) {
            const int tap = 2 * c + j;
            const int kt = tap / 9, kh = (tap % 9) / 3, kw = tap % 3;
            const int tt = t + kt - 1, hh = h + kh - 1, ww = wpix + kw - 1;
            const bool ok = (tap < NTAP) && ((unsigned)tt < T_) &&
                            ((unsigned)hh < H_) && ((unsigned)ww < W_);
            const float* base = Xb + (long)(8 * g) * THW_ + (long)tt * (H_ * W_) + hh * W_ + ww;
            #pragma unroll
            for (int q = 0; q < 4; ++q) {
                float f0 = ok ? base[(long)(2 * q + 0) * THW_] : 0.0f;
                float f1 = ok ? base[(long)(2 * q + 1) * THW_] : 0.0f;
                uint32_t u0 = __float_as_uint(f0), u1 = __float_as_uint(f1);
                float l0 = f0 - __uint_as_float(u0 & 0xFFFF0000u);
                float l1 = f1 - __uint_as_float(u1 & 0xFFFF0000u);
                ah[j*4+q] = (u0 >> 16) | (u1 & 0xFFFF0000u);
                al[j*4+q] = (__float_as_uint(l0) >> 16) | (__float_as_uint(l1) & 0xFFFF0000u);
            }
        }
        bf16x16 A_hi = __builtin_bit_cast(bf16x16, ah);
        bf16x16 A_lo = __builtin_bit_cast(bf16x16, al);
        const int tap0 = 2 * c, tap1 = 2 * c + 1;
        auto loadB = [&](int hl, int tn) -> bf16x16 {
            int n = tn * 16 + col;
            u32x4 q0 = *(const u32x4*)&Wl[(((hl * NTAPP) + tap0) * CO_ + n) * CI_ + 8 * g];
            u32x4 q1 = *(const u32x4*)&Wl[(((hl * NTAPP) + tap1) * CO_ + n) * CI_ + 8 * g];
            u32x8 bw;
            #pragma unroll
            for (int q = 0; q < 4; ++q) { bw[q] = q0[q]; bw[4+q] = q1[q]; }
            return __builtin_bit_cast(bf16x16, bw);
        };
        bf16x16 Bh0 = loadB(0, 0), Bl0 = loadB(1, 0);
        bf16x16 Bh1 = loadB(0, 1), Bl1 = loadB(1, 1);
        acc0 = __builtin_amdgcn_wmma_f32_16x16x32_bf16(false, A_hi, false, Bh0, (short)0, acc0, false, false);
        acc0 = __builtin_amdgcn_wmma_f32_16x16x32_bf16(false, A_lo, false, Bh0, (short)0, acc0, false, false);
        acc0 = __builtin_amdgcn_wmma_f32_16x16x32_bf16(false, A_hi, false, Bl0, (short)0, acc0, false, false);
        acc1 = __builtin_amdgcn_wmma_f32_16x16x32_bf16(false, A_hi, false, Bh1, (short)0, acc1, false, false);
        acc1 = __builtin_amdgcn_wmma_f32_16x16x32_bf16(false, A_lo, false, Bh1, (short)0, acc1, false, false);
        acc1 = __builtin_amdgcn_wmma_f32_16x16x32_bf16(false, A_hi, false, Bl1, (short)0, acc1, false, false);
    }

    __syncthreads();
    #pragma unroll
    for (int j = 0; j < 8; ++j) {
        int w = wbase + j + 8 * g;
        Ot[col * 129 + w]        = acc0[j];
        Ot[(16 + col) * 129 + w] = acc1[j];
    }
    __syncthreads();
    const long oB = (((long)b * CO_) * T_ + t) * (long)(H_ * W_) + (long)h * W_;
    for (int i = tid; i < CO_ * W_; i += 256) {
        int n = i >> 7, w = i & 127;
        Out[oB + (long)n * THW_ + w] = Ot[n * 129 + w] + Bias[n];
    }
}

extern "C" void kernel_launch(void* const* d_in, const int* in_sizes, int n_in,
                              void* d_out, int out_size, void* d_ws, size_t ws_size,
                              hipStream_t stream) {
    (void)in_sizes; (void)n_in; (void)out_size;
    const float* x    = (const float*)d_in[0];
    const float* w    = (const float*)d_in[1];
    const float* bias = (const float*)d_in[2];
    float* out = (float*)d_out;

    const size_t planeElems = (size_t)PPIX * CI_;
    const size_t need = 2ull * planeElems * sizeof(unsigned short);  // ~73.5 MB
    if (ws_size >= need) {
        unsigned short* wshi = (unsigned short*)d_ws;
        unsigned short* wslo = wshi + planeElems;
        split_pad_bf16_kernel<<<(PPIX + 255) / 256, 256, 0, stream>>>(x, wshi, wslo);
        conv3d_wmma_ws_kernel<<<B_ * T_ * H_, 256, 0, stream>>>(wshi, wslo, w, bias, out);
    } else {
        conv3d_wmma_fallback_kernel<<<B_ * T_ * H_, 256, 0, stream>>>(x, w, bias, out);
    }
}
